// Attention_44341242364527
// MI455X (gfx1250) — compile-verified, matrix operand order checked by reading, not run on hardware
//
#include <hip/hip_runtime.h>
#include <hip/hip_bf16.h>
#include <stdint.h>

// ---------------- Problem constants (B=1) ----------------
constexpr int S    = 2048;
constexpr int H    = 16;
constexpr int D    = 128;
constexpr int KSEL = 64;
constexpr int HID  = H * D;      // 2048

// ---------------- WMMA vector types ----------------
typedef __bf16 bf16_t;
typedef __attribute__((ext_vector_type(16))) __bf16 v16bf;
typedef __attribute__((ext_vector_type(8)))  __bf16 v8bf;
typedef __attribute__((ext_vector_type(8)))  float  v8f;

// ---------------- CDNA5 async global->LDS copy (ASYNCcnt-tracked) ----------------
// global_load_async_to_lds_b128: LDS[vdst_lane] = MEM[vaddr_lane], 16 bytes, no VGPR round trip.
__device__ __forceinline__ void async_copy_b128(uint32_t lds_off, const void* gptr) {
    asm volatile("global_load_async_to_lds_b128 %0, %1, off"
                 :: "v"(lds_off), "v"((uint64_t)(uintptr_t)gptr)
                 : "memory");
}
__device__ __forceinline__ void wait_async_0() {
    asm volatile("s_wait_asynccnt 0x0" ::: "memory");
}
__device__ __forceinline__ void wait_async_8() {   // allow 8 outstanding (next tile in flight)
    asm volatile("s_wait_asynccnt 0x8" ::: "memory");
}

// ---------------- fp32 -> bf16 cast ----------------
__global__ __launch_bounds__(256)
void f32_to_bf16_kernel(const float* __restrict__ in, bf16_t* __restrict__ out, size_t n) {
    size_t i      = (size_t)blockIdx.x * blockDim.x + threadIdx.x;
    size_t stride = (size_t)gridDim.x * blockDim.x;
    for (; i < n; i += stride) out[i] = (bf16_t)in[i];
}

// ---------------- bf16 WMMA GEMM: C[M,N] = X[M,K] * W[N,K]^T (+bias) ----------------
// Block: 128 threads = 4 waves in a 2x2 grid; block tile 128x128, K-step 32.
// Each wave owns a 64x64 patch = 4x4 WMMA tiles (16x v_wmma_f32_16x16x32_bf16 per K-step).
// Double-buffered LDS fed by async global->LDS copies (8 issues/thread/buffer).
constexpr int BM = 128, BN = 128, BK = 32;
constexpr int LDST = 48;            // padded LDS row stride (elems); 96B, 16B aligned
constexpr int GEMM_THREADS = 128;

__global__ __launch_bounds__(GEMM_THREADS)
void gemm_bf16_wmma(const bf16_t* __restrict__ X, const bf16_t* __restrict__ W,
                    const float* __restrict__ bias, float* __restrict__ Cout,
                    int Mdim, int Ndim, int Kdim)
{
    __shared__ bf16_t Xs[2][BM * LDST];
    __shared__ bf16_t Ws[2][BN * LDST];

    const int tid  = threadIdx.x;
    const int wave = tid >> 5;
    const int lane = tid & 31;
    const int m0   = blockIdx.y * BM;
    const int n0   = blockIdx.x * BN;
    const int wm   = (wave >> 1) * 64;   // wave row offset in block tile
    const int wn   = (wave & 1)  * 64;   // wave col offset in block tile

    v8f acc[4][4] = {};                  // fp32 accumulators: 4x4 16x16 tiles

    const int lrow  = lane & 15;         // row (A) / col (B) owned by this lane
    const int khalf = (lane >> 4) * 8;   // K sub-block per ISA 16-bit A/B layout

    // cooperative async stage of one K-tile into buffer `buf` (8 b128 issues per thread)
    auto stage = [&](int buf, int k0) {
        #pragma unroll
        for (int p = 0; p < 4; ++p) {              // X tile: 128x32 bf16
            int linear = (p * GEMM_THREADS + tid) * 8;
            int r = linear >> 5;
            int c = linear & 31;
            async_copy_b128((uint32_t)(uintptr_t)&Xs[buf][r * LDST + c],
                            &X[(size_t)(m0 + r) * Kdim + k0 + c]);
        }
        #pragma unroll
        for (int p = 0; p < 4; ++p) {              // W tile: 128x32 bf16
            int linear = (p * GEMM_THREADS + tid) * 8;
            int r = linear >> 5;
            int c = linear & 31;
            async_copy_b128((uint32_t)(uintptr_t)&Ws[buf][r * LDST + c],
                            &W[(size_t)(n0 + r) * Kdim + k0 + c]);
        }
    };

    stage(0, 0);
    const int nsteps = Kdim / BK;
    for (int step = 0; step < nsteps; ++step) {
        const int buf = step & 1;
        if (step + 1 < nsteps) {
            stage(buf ^ 1, (step + 1) * BK);       // prefetch next tile into other buffer
            wait_async_8();                        // current tile's 8 issues complete
        } else {
            wait_async_0();
        }
        __syncthreads();                           // all waves' LDS data visible

        // build fragments from LDS (two 16B ds reads each, ISA Sec 7.12.2 layout)
        v16bf afrag[4], bfrag[4];
        #pragma unroll
        for (int i = 0; i < 4; ++i) {
            const bf16_t* base = &Xs[buf][(wm + i * 16 + lrow) * LDST + khalf];
            v8bf lo = *(const v8bf*)base;
            v8bf hi = *(const v8bf*)(base + 16);
            afrag[i] = __builtin_shufflevector(lo, hi,
                0,1,2,3,4,5,6,7,8,9,10,11,12,13,14,15);
        }
        #pragma unroll
        for (int j = 0; j < 4; ++j) {
            const bf16_t* base = &Ws[buf][(wn + j * 16 + lrow) * LDST + khalf];
            v8bf lo = *(const v8bf*)base;
            v8bf hi = *(const v8bf*)(base + 16);
            bfrag[j] = __builtin_shufflevector(lo, hi,
                0,1,2,3,4,5,6,7,8,9,10,11,12,13,14,15);
        }

        // 16x v_wmma_f32_16x16x32_bf16
        #pragma unroll
        for (int i = 0; i < 4; ++i)
            #pragma unroll
            for (int j = 0; j < 4; ++j)
                acc[i][j] = __builtin_amdgcn_wmma_f32_16x16x32_bf16(
                    false, afrag[i], false, bfrag[j],
                    (short)0, acc[i][j], false, false);

        __syncthreads();                           // done reading `buf` before reuse
    }

    // epilogue: C VGPR r holds row (r + 8*(lane>>4)), col (lane&15)
    const int crow = (lane >> 4) * 8;
    const int ccol = lane & 15;
    #pragma unroll
    for (int i = 0; i < 4; ++i) {
        #pragma unroll
        for (int j = 0; j < 4; ++j) {
            int n = n0 + wn + j * 16 + ccol;
            float b = bias ? bias[n] : 0.0f;
            #pragma unroll
            for (int r = 0; r < 8; ++r) {
                int m = m0 + wm + i * 16 + crow + r;
                Cout[(size_t)m * Ndim + n] = acc[i][j][r] + b;
            }
        }
    }
}

// ---------------- gathered attention core: one wave per (h, s) ----------------
__global__ __launch_bounds__(256)
void attn_core(const float* __restrict__ Q, const float* __restrict__ Km,
               const float* __restrict__ Vm, const int* __restrict__ idx,
               const unsigned char* __restrict__ valid,
               const float* __restrict__ geo, float* __restrict__ out)
{
    const int wid  = blockIdx.x * 8 + (threadIdx.x >> 5);
    const int lane = threadIdx.x & 31;
    const int h = wid & (H - 1);
    const int s = wid >> 4;          // / H
    if (s >= S) return;

    const float scale = 0.088388347648318447f;   // 1/sqrt(D), D=128
    const float* qrow = Q + (size_t)s * HID + h * D;

    // two K-slots per lane: k0 = lane, k1 = lane+32
    const int kA = lane, kB = lane + 32;
    const int iA = idx[(size_t)s * KSEL + kA];
    const int iB = idx[(size_t)s * KSEL + kB];
    const float* krowA = Km + (size_t)iA * HID + h * D;
    const float* krowB = Km + (size_t)iB * HID + h * D;

    float dotA = 0.0f, dotB = 0.0f;
    #pragma unroll
    for (int c = 0; c < D / 4; ++c) {
        float4 q4 = *(const float4*)(qrow  + 4 * c);   // uniform address -> broadcast
        float4 ka = *(const float4*)(krowA + 4 * c);
        float4 kb = *(const float4*)(krowB + 4 * c);
        dotA += q4.x * ka.x + q4.y * ka.y + q4.z * ka.z + q4.w * ka.w;
        dotB += q4.x * kb.x + q4.y * kb.y + q4.z * kb.z + q4.w * kb.w;
    }

    const size_t gbase = ((size_t)h * S + s) * KSEL;
    float lA = dotA * scale + geo[gbase + kA];
    float lB = dotB * scale + geo[gbase + kB];
    const bool vA = valid[(size_t)s * KSEL + kA] && (iA <= s);
    const bool vB = valid[(size_t)s * KSEL + kB] && (iB <= s);
    if (!vA) lA = -__builtin_inff();
    if (!vB) lB = -__builtin_inff();

    // softmax over 64 logits (2 per lane) via wave32 xor-shuffles
    float m = fmaxf(lA, lB);
    #pragma unroll
    for (int off = 16; off > 0; off >>= 1)
        m = fmaxf(m, __shfl_xor(m, off, 32));
    float eA = vA ? __expf(lA - m) : 0.0f;
    float eB = vB ? __expf(lB - m) : 0.0f;
    float sum = eA + eB;
    #pragma unroll
    for (int off = 16; off > 0; off >>= 1)
        sum += __shfl_xor(sum, off, 32);
    const float inv = 1.0f / sum;
    const float aA = eA * inv, aB = eB * inv;

    // out[d] = sum_k attn[k] * V[idx[k]][h*D + d]; lane owns d = lane + 32j
    float acc[4] = {0.f, 0.f, 0.f, 0.f};
    for (int k = 0; k < 32; ++k) {
        float w  = __shfl(aA, k, 32);
        int   kv = __shfl(iA, k, 32);
        const float* vrow = Vm + (size_t)kv * HID + h * D;
        #pragma unroll
        for (int j = 0; j < 4; ++j) acc[j] += w * vrow[lane + 32 * j];
    }
    for (int k = 0; k < 32; ++k) {
        float w  = __shfl(aB, k, 32);
        int   kv = __shfl(iB, k, 32);
        const float* vrow = Vm + (size_t)kv * HID + h * D;
        #pragma unroll
        for (int j = 0; j < 4; ++j) acc[j] += w * vrow[lane + 32 * j];
    }

    float* orow = out + (size_t)s * HID + h * D;
    #pragma unroll
    for (int j = 0; j < 4; ++j) orow[lane + 32 * j] = acc[j];
}

// ---------------- driver ----------------
extern "C" void kernel_launch(void* const* d_in, const int* in_sizes, int n_in,
                              void* d_out, int out_size, void* d_ws, size_t ws_size,
                              hipStream_t stream)
{
    const float*         hidden = (const float*)d_in[0];           // (S, HID)
    const int*           idx    = (const int*)d_in[1];             // (S, KSEL) int32
    const unsigned char* valid  = (const unsigned char*)d_in[2];   // (S, KSEL) bool
    const float*         geo    = (const float*)d_in[3];           // (H, S, KSEL)
    const float*         Wq     = (const float*)d_in[4];           // (HID, HID)
    const float*         Wk     = (const float*)d_in[5];
    const float*         Wv     = (const float*)d_in[6];
    const float*         Wo     = (const float*)d_in[7];
    const float*         bo     = (const float*)d_in[8];           // (HID)
    float*               out    = (float*)d_out;                   // (S, HID)

    const size_t NE = (size_t)S * HID;     // 4M elements
    char* ws = (char*)d_ws;
    bf16_t* Xb   = (bf16_t*)(ws);                    // 8 MB each bf16 buffer
    bf16_t* Wqb  = (bf16_t*)(ws + 1 * NE * 2);
    bf16_t* Wkb  = (bf16_t*)(ws + 2 * NE * 2);
    bf16_t* Wvb  = (bf16_t*)(ws + 3 * NE * 2);
    bf16_t* Wob  = (bf16_t*)(ws + 4 * NE * 2);
    float*  Qf   = (float*)(ws + 5 * NE * 2);        // 16 MB each fp32 buffer
    float*  Kf   = (float*)(ws + 5 * NE * 2 + 1 * NE * 4);
    float*  Vf   = (float*)(ws + 5 * NE * 2 + 2 * NE * 4);
    float*  AOf  = (float*)(ws + 5 * NE * 2 + 3 * NE * 4);
    bf16_t* AOb  = (bf16_t*)(ws + 5 * NE * 2 + 4 * NE * 4);

    // 1) casts to bf16
    f32_to_bf16_kernel<<<1024, 256, 0, stream>>>(hidden, Xb,  NE);
    f32_to_bf16_kernel<<<1024, 256, 0, stream>>>(Wq,     Wqb, NE);
    f32_to_bf16_kernel<<<1024, 256, 0, stream>>>(Wk,     Wkb, NE);
    f32_to_bf16_kernel<<<1024, 256, 0, stream>>>(Wv,     Wvb, NE);
    f32_to_bf16_kernel<<<1024, 256, 0, stream>>>(Wo,     Wob, NE);

    // 2) Q/K/V projections (WMMA bf16, async-LDS double-buffered)
    dim3 ggrid(HID / BN, S / BM);   // (16, 16)
    gemm_bf16_wmma<<<ggrid, GEMM_THREADS, 0, stream>>>(Xb, Wqb, nullptr, Qf, S, HID, HID);
    gemm_bf16_wmma<<<ggrid, GEMM_THREADS, 0, stream>>>(Xb, Wkb, nullptr, Kf, S, HID, HID);
    gemm_bf16_wmma<<<ggrid, GEMM_THREADS, 0, stream>>>(Xb, Wvb, nullptr, Vf, S, HID, HID);

    // 3) gathered attention core: one wave per (h, s)
    attn_core<<<(H * S) / 8, 256, 0, stream>>>(Qf, Kf, Vf, idx, valid, geo, AOf);

    // 4) output projection with bias
    f32_to_bf16_kernel<<<1024, 256, 0, stream>>>(AOf, AOb, NE);
    gemm_bf16_wmma<<<ggrid, GEMM_THREADS, 0, stream>>>(AOb, Wob, bo, out, S, HID, HID);
}
